// NNUE_65841848648290
// MI455X (gfx1250) — matrix operand order repaired and checked by reading, not compiled, not run on hardware
//
#include <hip/hip_runtime.h>

// ---------------------------------------------------------------------------
// NNUE forward for MI455X (gfx1250, wave32).
//  - Feature transformer: bandwidth-bound sparse scan (masks are ~30/41024
//    dense ones). 1.34 GB of mask reads dominates -> ~58us floor at 23.3 TB/s.
//  - Tail MLP: v_wmma_f32_16x16x32_f16 (f16 in, f32 accumulate) with
//    pre-swizzled B operands (2x global_load_b128 per operand per lane).
// ---------------------------------------------------------------------------

typedef __attribute__((ext_vector_type(16))) _Float16 v16h;
typedef __attribute__((ext_vector_type(8)))  _Float16 v8h;
typedef __attribute__((ext_vector_type(8)))  float    v8f;

__device__ __forceinline__ float clip01(float x) {
    return __builtin_fminf(__builtin_fmaxf(x, 0.0f), 1.0f);
}

// ---------------------------------------------------------------------------
// Kernel 0: convert W1 (256x32) and W2 (32x32) to f16, pre-swizzled into the
// WMMA B-operand lane layout (16-bit B 32x16, per ISA 7.12.2 / 7.12.4):
//   lane<16 : N=lane,    halves j=0..7 hold K = kb+j, j=8..15 hold K = kb+8+j (kb = ktile*32)
//   lane>=16: N=lane-16, halves j=0..7 hold K = kb+8+j, j=8..15 hold K = kb+16+j
// Storage: W1s[((ktile*2 + ntile)*32 + lane)*16 + j]  (16 halves = 32B/lane,
// consecutive lanes contiguous -> tail kernel loads 2x b128 per operand).
// ---------------------------------------------------------------------------
__global__ void nnue_cvt_kernel(const float* __restrict__ W1,
                                const float* __restrict__ W2,
                                _Float16* __restrict__ W1s,
                                _Float16* __restrict__ W2s) {
    int o = blockIdx.x * blockDim.x + threadIdx.x;
    if (o < 8192) {                       // W1: 8 ktiles x 2 ntiles x 32 lanes x 16
        const int j     = o & 15;
        const int lane  = (o >> 4) & 31;
        const int ntile = (o >> 9) & 1;
        const int kt    = o >> 10;
        const int k = kt * 32 + (j >> 3) * 16 + (lane >> 4) * 8 + (j & 7);
        const int n = ntile * 16 + (lane & 15);
        W1s[o] = (_Float16)W1[k * 32 + n];
    } else if (o < 8192 + 1024) {         // W2: 1 ktile x 2 ntiles x 32 lanes x 16
        const int p     = o - 8192;
        const int j     = p & 15;
        const int lane  = (p >> 4) & 31;
        const int ntile = p >> 9;
        const int k = (j >> 3) * 16 + (lane >> 4) * 8 + (j & 7);
        const int n = ntile * 16 + (lane & 15);
        W2s[p] = (_Float16)W2[k * 32 + n];
    }
}

// ---------------------------------------------------------------------------
// Kernel 1: feature transformer. One 128-thread block per batch row.
// Phase A: strided float4 scan of the 41024-wide 0/1 mask, buffering nonzero
//          feature indices per-thread (deterministic order).
// Phase B: LDS prefix-sum compaction (no atomics -> deterministic sum order),
//          then coalesced accumulation of W_ft rows (thread t = column t).
// Then apply us/them perspective concat + clip, store l0 as f16.
// ---------------------------------------------------------------------------
__global__ __launch_bounds__(128) void nnue_ft_kernel(
    const float* __restrict__ us,   const float* __restrict__ them,
    const float* __restrict__ w_in, const float* __restrict__ b_in,
    const float* __restrict__ W_ft, const float* __restrict__ b_ft,
    _Float16* __restrict__ l0h)
{
    const int row = blockIdx.x;
    const int tid = threadIdx.x;

    __shared__ int cnt_s[128];
    __shared__ int idxs[512];

    float acc0 = 0.0f, acc1 = 0.0f;

    for (int persp = 0; persp < 2; ++persp) {
        const float* mask = (persp ? b_in : w_in) + (size_t)row * 41024;
        const int4* m4 = (const int4*)mask;       // 41024/4 = 10256 vec4 loads

        int loc[12];
        int c = 0;
        for (int j = tid; j < 10256; j += 128) {
            __builtin_prefetch(&m4[j + 512], 0, 0);   // global_prefetch_b8
            int4 v = m4[j];                            // 0.0f bits == 0
            if ((v.x | v.y | v.z | v.w) != 0) {
                if (v.x != 0) { if (c < 12) loc[c] = 4 * j + 0; ++c; }
                if (v.y != 0) { if (c < 12) loc[c] = 4 * j + 1; ++c; }
                if (v.z != 0) { if (c < 12) loc[c] = 4 * j + 2; ++c; }
                if (v.w != 0) { if (c < 12) loc[c] = 4 * j + 3; ++c; }
            }
        }
        int cc = (c > 12) ? 12 : c;
        cnt_s[tid] = cc;
        __syncthreads();

        int off = 0, total = 0;
        for (int i = 0; i < 128; ++i) {
            int v = cnt_s[i];
            total += v;
            if (i < tid) off += v;
        }
        if (total > 512) total = 512;
        for (int i = 0; i < cc; ++i) {
            int p = off + i;
            if (p < 512) idxs[p] = loc[i];
        }
        __syncthreads();

        float a = b_ft[tid];
        for (int j = 0; j < total; ++j) {
            a += W_ft[(size_t)idxs[j] * 128 + tid];   // 512B coalesced, L2-hit
        }
        if (persp == 0) acc0 = a; else acc1 = a;
        __syncthreads();
    }

    const float u = us[row];
    const float t = them[row];
    const float fw = acc0, fb = acc1;
    l0h[(size_t)row * 256 + tid]       = (_Float16)clip01(u * fw + t * fb);
    l0h[(size_t)row * 256 + 128 + tid] = (_Float16)clip01(u * fb + t * fw);
}

// ---------------------------------------------------------------------------
// Kernel 2: tail MLP with WMMA. One wave (32 threads) per 16-row tile.
//   l1 = clip(l0 @ W1 + b1)   : 16 x v_wmma_f32_16x16x32_f16 (K=256, 2 N-tiles)
//   l2 = clip(l1 @ W2 + b2)   : 2  x v_wmma_f32_16x16x32_f16 (K=32)
//   out = l2 @ Wo + bo        : 32-wide dot from LDS
// A-operand layout (16-bit A 16x32, per ISA 7.12.2):
//   lane<16 : M=lane,   VGPR0..3 = K 0..7,   VGPR4..7 = K 16..23
//   lane>=16: M=lane-16,VGPR0..3 = K 8..15,  VGPR4..7 = K 24..31
// -> two b128 loads per lane from row-major f16.
// B operands come pre-swizzled (see nnue_cvt_kernel) -> two b128 loads each.
// ---------------------------------------------------------------------------
__global__ __launch_bounds__(32) void nnue_tail_kernel(
    const _Float16* __restrict__ l0h,
    const _Float16* __restrict__ W1s, const float* __restrict__ b1,
    const _Float16* __restrict__ W2s, const float* __restrict__ b2,
    const float* __restrict__ Wo,     const float* __restrict__ bo,
    float* __restrict__ out)
{
    const int lane = threadIdx.x;
    const int ln = lane & 15;
    const int hs = lane >> 4;          // 0 = low half lanes, 1 = high half
    const int m0 = blockIdx.x * 16;

    __shared__ _Float16 lds_h[16 * 32];
    __shared__ float    lds_f[16 * 32];

    union AV { v16h v; v8h h8[2]; _Float16 h[16]; };

    // ---- layer 1: C init = bias broadcast down M ----
    v8f c0, c1;
    {
        const float bb0 = b1[ln];
        const float bb1 = b1[ln + 16];
        for (int i = 0; i < 8; ++i) { c0[i] = bb0; c1[i] = bb1; }
    }
    const int mrow = m0 + ln;
    for (int ks = 0; ks < 8; ++ks) {
        const int k0 = ks * 32;
        AV a;
        const _Float16* ap = l0h + (size_t)mrow * 256 + k0 + hs * 8;
        a.h8[0] = *(const v8h*)(ap);        // K k0+{0..7}  (or +8..15)
        a.h8[1] = *(const v8h*)(ap + 16);   // K k0+{16..23}(or +24..31)

        AV b0v, b1v;
        const v8h* p0 = (const v8h*)(W1s + ((size_t)((ks * 2 + 0) * 32 + lane) * 16));
        const v8h* p1 = (const v8h*)(W1s + ((size_t)((ks * 2 + 1) * 32 + lane) * 16));
        b0v.h8[0] = p0[0]; b0v.h8[1] = p0[1];
        b1v.h8[0] = p1[0]; b1v.h8[1] = p1[1];

        c0 = __builtin_amdgcn_wmma_f32_16x16x32_f16(false, a.v, false, b0v.v,
                                                    (short)0, c0, false, false);
        c1 = __builtin_amdgcn_wmma_f32_16x16x32_f16(false, a.v, false, b1v.v,
                                                    (short)0, c1, false, false);
    }

    // ---- l1 tile (clipped) -> LDS f16 row-major [16][32] ----
    for (int v = 0; v < 8; ++v) {
        const int m = v + 8 * hs;       // C/D layout: lanes16-31 hold M=8..15
        lds_h[m * 32 + ln]      = (_Float16)clip01(c0[v]);
        lds_h[m * 32 + 16 + ln] = (_Float16)clip01(c1[v]);
    }
    __syncthreads();

    // ---- layer 2 ----
    AV a2;
    {
        const _Float16* ap = &lds_h[ln * 32 + hs * 8];
        a2.h8[0] = *(const v8h*)(ap);
        a2.h8[1] = *(const v8h*)(ap + 16);
    }
    AV b20, b21;
    {
        const v8h* p0 = (const v8h*)(W2s + ((size_t)(0 * 32 + lane) * 16));
        const v8h* p1 = (const v8h*)(W2s + ((size_t)(1 * 32 + lane) * 16));
        b20.h8[0] = p0[0]; b20.h8[1] = p0[1];
        b21.h8[0] = p1[0]; b21.h8[1] = p1[1];
    }
    v8f d0, d1;
    {
        const float bb0 = b2[ln];
        const float bb1 = b2[ln + 16];
        for (int i = 0; i < 8; ++i) { d0[i] = bb0; d1[i] = bb1; }
    }
    d0 = __builtin_amdgcn_wmma_f32_16x16x32_f16(false, a2.v, false, b20.v,
                                                (short)0, d0, false, false);
    d1 = __builtin_amdgcn_wmma_f32_16x16x32_f16(false, a2.v, false, b21.v,
                                                (short)0, d1, false, false);

    for (int v = 0; v < 8; ++v) {
        const int m = v + 8 * hs;
        lds_f[m * 32 + ln]      = clip01(d0[v]);
        lds_f[m * 32 + 16 + ln] = clip01(d1[v]);
    }
    __syncthreads();

    // ---- output layer: per-row 32-wide dot ----
    if (lane < 16) {
        float s = bo[0];
        for (int n = 0; n < 32; ++n) s += lds_f[lane * 32 + n] * Wo[n];
        out[m0 + lane] = s;
    }
}

// ---------------------------------------------------------------------------
extern "C" void kernel_launch(void* const* d_in, const int* in_sizes, int n_in,
                              void* d_out, int out_size, void* d_ws, size_t ws_size,
                              hipStream_t stream) {
    const float* us   = (const float*)d_in[0];
    const float* them = (const float*)d_in[1];
    const float* w_in = (const float*)d_in[2];
    const float* b_in = (const float*)d_in[3];
    const float* W_ft = (const float*)d_in[4];
    const float* b_ft = (const float*)d_in[5];
    const float* W1   = (const float*)d_in[6];
    const float* b1   = (const float*)d_in[7];
    const float* W2   = (const float*)d_in[8];
    const float* b2   = (const float*)d_in[9];
    const float* Wo   = (const float*)d_in[10];
    const float* bo   = (const float*)d_in[11];
    float* out = (float*)d_out;

    const int B = in_sizes[0];            // 4096

    // workspace layout (needs ~2.02 MB):
    //   [0, 16384)        W1 swizzled f16 (256*32)
    //   [16384, 18432)    W2 swizzled f16 (32*32)
    //   [18432, ...)      l0 as f16 (B*256), 16B-aligned for b128 loads
    _Float16* W1s = (_Float16*)d_ws;
    _Float16* W2s = (_Float16*)((char*)d_ws + 16384);
    _Float16* l0h = (_Float16*)((char*)d_ws + 18432);

    nnue_cvt_kernel<<<36, 256, 0, stream>>>(W1, W2, W1s, W2s);
    nnue_ft_kernel<<<B, 128, 0, stream>>>(us, them, w_in, b_in, W_ft, b_ft, l0h);
    nnue_tail_kernel<<<B / 16, 32, 0, stream>>>(l0h, W1s, b1, W2s, b2, Wo, bo, out);
}
